// MultiHeadAttention_81415400063598
// MI455X (gfx1250) — compile-verified, matrix operand order checked by reading, not run on hardware
//
#include <hip/hip_runtime.h>
#include <hip/hip_bf16.h>
#include <math.h>

// Problem constants (match reference)
#define B_    2
#define S_    2048
#define DIM_  1024
#define H_    16
#define DH_   64
#define MROWS (B_ * S_)   // 4096

typedef __attribute__((ext_vector_type(16))) __bf16 v16bf;
typedef __attribute__((ext_vector_type(8)))  __bf16 v8bf;
typedef __attribute__((ext_vector_type(8)))  float  v8f;
typedef __attribute__((ext_vector_type(4)))  unsigned int u32x4;
typedef __attribute__((ext_vector_type(8)))  int i32x8;
typedef __attribute__((ext_vector_type(4)))  int i32x4;

static __device__ __forceinline__ v16bf frag_cat(v8bf lo, v8bf hi) {
  return __builtin_shufflevector(lo, hi, 0,1,2,3,4,5,6,7,8,9,10,11,12,13,14,15);
}

// Load a 16-bit A/B fragment for one lane from a row-major (contiguous-K) row.
// Lane half (hl) selects K octets {hl*8..hl*8+7} and {16+hl*8..16+hl*8+7}
// per the CDNA5 16-bit 16x32 matrix VGPR layout.
static __device__ __forceinline__ v16bf load_frag(const __bf16* row, int k0, int hl) {
  v8bf lo = *(const v8bf*)(row + k0 + hl * 8);
  v8bf hi = *(const v8bf*)(row + k0 + 16 + hl * 8);
  return frag_cat(lo, hi);
}

static __device__ __forceinline__ v8f wmma_bf16(v16bf a, v16bf b, v8f c) {
  return __builtin_amdgcn_wmma_f32_16x16x32_bf16(false, a, false, b, (short)0, c,
                                                 false, false);
}

// ---------------------------------------------------------------------------
// TDM: async-DMA a 2D tile (64 rows x 32 cols, bf16, row stride S_) from
// global memory into LDS. Tracked by TENSORcnt. D# per CDNA5 ISA 8.3/8.4.
// This toolchain exposes the 6-arg builtin:
//   (u32x4 g0, i32x8 g1, i32x4 g2, i32x4 g3, i32x8 pad, i32 cpol)
static __device__ __forceinline__ void tdm_load_v_tile(const __bf16* gsrc,
                                                       __bf16* ldst) {
  const unsigned long long ga = (unsigned long long)(uintptr_t)gsrc;
  const unsigned int la = (unsigned int)(uintptr_t)ldst;   // LDS byte address
  u32x4 g0;
  g0[0] = 1u;                                            // count=1, user D#
  g0[1] = la;                                            // lds_addr
  g0[2] = (unsigned int)ga;                              // global_addr[31:0]
  g0[3] = (unsigned int)((ga >> 32) & 0x01FFFFFFu) | (2u << 30);  // type=2
  i32x8 g1;
  g1[0] = (int)(1u << 16);          // data_size = 2 bytes; no multicast/pad
  g1[1] = (int)(32u << 16);         // tensor_dim0 = 32 (keys, contiguous)
  g1[2] = (int)(64u << 16);         // tensor_dim0[31:16]=0 | tensor_dim1 = 64
  g1[3] = (int)(32u << 16);         // tensor_dim1[31:16]=0 | tile_dim0 = 32
  g1[4] = 64;                       // tile_dim1 = 64 | tile_dim2 = 0
  g1[5] = (int)S_;                  // tensor_dim0_stride = 2048 elements
  g1[6] = 0;                        // stride0[47:32]=0 | stride1[15:0]=0
  g1[7] = 0;
  const i32x4 z4 = {0, 0, 0, 0};    // groups 2/3 unused (2D tensor)
  const i32x8 z8 = {0, 0, 0, 0, 0, 0, 0, 0};
  __builtin_amdgcn_tensor_load_to_lds(g0, g1, z4, z4, z8, 0);
}

// ---------------------------------------------------------------------------
// Cast fp32 activations -> bf16 (row-major, [MROWS, DIM])
__global__ __launch_bounds__(256)
void cast_in_kernel(const float* __restrict__ q, const float* __restrict__ k,
                    const float* __restrict__ v, __bf16* __restrict__ xq,
                    __bf16* __restrict__ xk, __bf16* __restrict__ xv) {
  const size_t i = (size_t)blockIdx.x * 256 + threadIdx.x;
  xq[i] = (__bf16)q[i];
  xk[i] = (__bf16)k[i];
  xv[i] = (__bf16)v[i];
}

// Cast + transpose weights: W[K,N] fp32 -> Wt[N,K] bf16 (contiguous-K rows)
__global__ __launch_bounds__(256)
void cast_w_kernel(const float* __restrict__ wq, const float* __restrict__ wk,
                   const float* __restrict__ wv, const float* __restrict__ wf,
                   __bf16* __restrict__ tq, __bf16* __restrict__ tk,
                   __bf16* __restrict__ tv, __bf16* __restrict__ tf) {
  const size_t i = (size_t)blockIdx.x * 256 + threadIdx.x;   // [0, 4*1M)
  const int sel = (int)(i >> 20);
  const int idx = (int)(i & ((1u << 20) - 1));
  const int n = idx >> 10;
  const int k = idx & 1023;
  const float* W = sel == 0 ? wq : sel == 1 ? wk : sel == 2 ? wv : wf;
  __bf16* T      = sel == 0 ? tq : sel == 1 ? tk : sel == 2 ? tv : tf;
  T[(size_t)n * DIM_ + k] = (__bf16)W[(size_t)k * DIM_ + n];
}

// ---------------------------------------------------------------------------
// WMMA GEMM: out = A[MROWS,DIM] (bf16) @ Wt^T (Wt is [N,K] bf16) + bias (f32)
// One wave computes a 64x64 macro-tile (4x4 WMMA tiles): 16 b128 loads per
// 16 WMMAs in the K-loop (1:1 load:math issue ratio).
// MODE 0: store bf16 head-split   [B,H,S,DH]   (Q, K)
// MODE 1: store bf16 head-split-T [B,H,DH,S]   (V transposed for PV B-frags)
// MODE 2: store fp32 row-major    [MROWS,DIM]  (final projection -> d_out)
template <int MODE>
__global__ __launch_bounds__(256)
void gemm_wmma_kernel(const __bf16* __restrict__ A, const __bf16* __restrict__ Wt,
                      const float* __restrict__ bias, void* __restrict__ outp) {
  const int lane = threadIdx.x & 31;
  const int hl = lane >> 4;          // lane half selects K octet
  const int lm = lane & 15;          // row/col within tile
  const int gw = blockIdx.x * 8 + (threadIdx.x >> 5);   // 1024 waves
  const int m0 = (gw >> 4) * 64;     // 64 M-macrotiles
  const int n0 = (gw & 15) * 64;     // 16 N-macrotiles

  v8f acc[4][4] = {};
  const __bf16* ar[4];
  const __bf16* br[4];
#pragma unroll
  for (int i = 0; i < 4; ++i) {
    ar[i] = A  + (size_t)(m0 + i * 16 + lm) * DIM_;
    br[i] = Wt + (size_t)(n0 + i * 16 + lm) * DIM_;
  }

  for (int k0 = 0; k0 < DIM_; k0 += 32) {
    v16bf af[4], bfv[4];
#pragma unroll
    for (int i = 0; i < 4; ++i) af[i]  = load_frag(ar[i], k0, hl);
#pragma unroll
    for (int j = 0; j < 4; ++j) bfv[j] = load_frag(br[j], k0, hl);
    if (k0 + 128 < DIM_) {
      // Warm WGP$/L2 for the K-block 4 iterations ahead (global_prefetch_b8).
      __builtin_prefetch(ar[0] + k0 + 128, 0, 1);
      __builtin_prefetch(br[0] + k0 + 128, 0, 1);
    }
#pragma unroll
    for (int i = 0; i < 4; ++i)
#pragma unroll
      for (int j = 0; j < 4; ++j)
        acc[i][j] = wmma_bf16(af[i], bfv[j], acc[i][j]);
  }

#pragma unroll
  for (int j = 0; j < 4; ++j) {
    const int n = n0 + j * 16 + lm;
    const float bn = bias[n];
    const int h = n >> 6, d = n & 63;
#pragma unroll
    for (int i = 0; i < 4; ++i) {
#pragma unroll
      for (int r = 0; r < 8; ++r) {
        // C/D layout: VGPR r holds row r (lanes 0-15) / row 8+r (lanes 16-31)
        const int m = m0 + i * 16 + r + 8 * hl;
        const float val = acc[i][j][r] + bn;
        const int b = m >> 11, s = m & (S_ - 1);
        if (MODE == 0) {
          ((__bf16*)outp)[((size_t)(b * H_ + h) * S_ + s) * DH_ + d] = (__bf16)val;
        } else if (MODE == 1) {
          ((__bf16*)outp)[((size_t)(b * H_ + h) * DH_ + d) * S_ + s] = (__bf16)val;
        } else {
          ((float*)outp)[(size_t)m * DIM_ + n] = val;
        }
      }
    }
  }
}

// ---------------------------------------------------------------------------
// Flash attention: one wave per (b, h, 16-query tile); online softmax over
// 32 keys per iteration. QK^T and PV on WMMA; V tiles streamed into LDS by
// the Tensor Data Mover (double-buffered, TENSORcnt); row sums via P@ones
// WMMA; row max via xor-shuffle over the wave32 halves.
__global__ __launch_bounds__(32)
void attn_kernel(const __bf16* __restrict__ Qh, const __bf16* __restrict__ Kh,
                 const __bf16* __restrict__ Vt, const int* __restrict__ pad_mask,
                 const int* __restrict__ train_p, __bf16* __restrict__ ctx) {
  const int lane = threadIdx.x & 31;
  const int hl = lane >> 4;
  const int lm = lane & 15;
  const int bh = blockIdx.x >> 7;      // 0..31  (b*H + h)
  const int qt = blockIdx.x & 127;     // query tile
  const int b = bh >> 4;
  const int s0 = qt * 16;

  const __bf16* qb = Qh + (size_t)bh * S_ * DH_;
  const __bf16* kb = Kh + (size_t)bh * S_ * DH_;
  const __bf16* vb = Vt + (size_t)bh * DH_ * S_;
  const int* maskb = pad_mask + b * S_;
  const int train = train_p[0];

  // Q fragments (A-matrix), rows = queries, K-dim = DH (two 32-chunks)
  const __bf16* qrow = qb + (size_t)(s0 + lm) * DH_;
  v16bf qf0 = load_frag(qrow, 0, hl);
  v16bf qf1 = load_frag(qrow, 32, hl);

  // All-ones B fragment: row-sum of P via one WMMA (lands in C layout).
  v16bf onesf;
#pragma unroll
  for (int i = 0; i < 16; ++i) onesf[i] = (__bf16)1.0f;

  float mrow[8], lrow[8];
#pragma unroll
  for (int r = 0; r < 8; ++r) { mrow[r] = -INFINITY; lrow[r] = 0.f; }
  v8f acc[4] = {};   // O tile [16 x 64] as 4 n-tiles

  __shared__ __align__(16) __bf16 pb[16 * 32];       // P re-layout (C -> A frag)
  __shared__ __align__(16) __bf16 vbuf[2][64 * 32];  // TDM V-tile double buffer

  // Kick off the first V tile DMA (rows = dh 0..63, cols = keys 0..31).
  tdm_load_v_tile(vb, &vbuf[0][0]);
  int cur = 0;

  const float scale = 1.0f / (float)DH_;   // reference divides by sqrt(DH) twice

  for (int j0 = 0; j0 < S_; j0 += 32) {
    // Prefetch next V tile into the other buffer while we compute this step.
    if (j0 + 32 < S_) tdm_load_v_tile(vb + j0 + 32, &vbuf[cur ^ 1][0]);

    // K fragments (B-matrix): lane = key column, contiguous over d
    const __bf16* kr0 = kb + (size_t)(j0 + lm) * DH_;
    const __bf16* kr1 = kr0 + (size_t)16 * DH_;
    v16bf k00 = load_frag(kr0, 0, hl), k01 = load_frag(kr0, 32, hl);
    v16bf k10 = load_frag(kr1, 0, hl), k11 = load_frag(kr1, 32, hl);

    v8f sc0 = {}, sc1 = {};
    sc0 = wmma_bf16(qf0, k00, sc0);
    sc0 = wmma_bf16(qf1, k01, sc0);
    sc1 = wmma_bf16(qf0, k10, sc1);
    sc1 = wmma_bf16(qf1, k11, sc1);

    const bool kill0 = train && (maskb[j0 + lm] == 0);
    const bool kill1 = train && (maskb[j0 + 16 + lm] == 0);
#pragma unroll
    for (int r = 0; r < 8; ++r) {
      sc0[r] = kill0 ? -1e9f : sc0[r] * scale;
      sc1[r] = kill1 ? -1e9f : sc1[r] * scale;
    }

    // Row max via xor-shuffle over the 16-lane half; rescale running state.
    float alpha[8];
#pragma unroll
    for (int r = 0; r < 8; ++r) {
      float v = fmaxf(sc0[r], sc1[r]);
      v = fmaxf(v, __shfl_xor(v, 1));
      v = fmaxf(v, __shfl_xor(v, 2));
      v = fmaxf(v, __shfl_xor(v, 4));
      v = fmaxf(v, __shfl_xor(v, 8));
      const float mn = fmaxf(mrow[r], v);
      alpha[r] = __expf(mrow[r] - mn);
      mrow[r] = mn;
      sc0[r] = __expf(sc0[r] - mn);
      sc1[r] = __expf(sc1[r] - mn);
    }
#pragma unroll
    for (int nt = 0; nt < 4; ++nt)
#pragma unroll
      for (int r = 0; r < 8; ++r) acc[nt][r] *= alpha[r];

    // P: C-layout -> A-fragment layout via LDS (single-wave workgroup).
    __syncthreads();
#pragma unroll
    for (int r = 0; r < 8; ++r) {
      const int m = r + 8 * hl;
      pb[m * 32 + lm]      = (__bf16)sc0[r];
      pb[m * 32 + 16 + lm] = (__bf16)sc1[r];
    }
    __syncthreads();
    v16bf pf = load_frag(pb + lm * 32, 0, hl);

    // Row sums of P with one WMMA against the all-ones matrix.
    v8f ls = {};
    ls = wmma_bf16(pf, onesf, ls);
#pragma unroll
    for (int r = 0; r < 8; ++r) lrow[r] = lrow[r] * alpha[r] + ls[r];

    // Consume the TDM-staged V tile (wait for its DMA; next one may fly).
    if (j0 + 32 < S_) __builtin_amdgcn_s_wait_tensorcnt(1);
    else              __builtin_amdgcn_s_wait_tensorcnt(0);
    __asm__ __volatile__("" ::: "memory");
#pragma unroll
    for (int nt = 0; nt < 4; ++nt) {
      const __bf16* vrow = &vbuf[cur][(nt * 16 + lm) * 32];
      v16bf vf = load_frag(vrow, 0, hl);
      acc[nt] = wmma_bf16(pf, vf, acc[nt]);
    }
    cur ^= 1;
  }

  // Normalize and store context as [B, S, DIM] bf16.
  const int h = bh & 15;
#pragma unroll
  for (int nt = 0; nt < 4; ++nt)
#pragma unroll
    for (int r = 0; r < 8; ++r) {
      const int m = r + 8 * hl;
      const int srow = s0 + m;
      const int col = h * DH_ + nt * 16 + lm;
      const float val = acc[nt][r] / lrow[r];
      ctx[((size_t)(b * S_ + srow)) * DIM_ + col] = (__bf16)val;
    }
}

// ---------------------------------------------------------------------------
extern "C" void kernel_launch(void* const* d_in, const int* in_sizes, int n_in,
                              void* d_out, int out_size, void* d_ws, size_t ws_size,
                              hipStream_t stream) {
  const float* q   = (const float*)d_in[0];
  const float* k   = (const float*)d_in[1];
  const float* v   = (const float*)d_in[2];
  const int* pad   = (const int*)d_in[3];
  const int* train = (const int*)d_in[4];
  const float* Wq  = (const float*)d_in[5];
  const float* bq  = (const float*)d_in[6];
  const float* Wk  = (const float*)d_in[7];
  const float* bk  = (const float*)d_in[8];
  const float* Wv  = (const float*)d_in[9];
  const float* bv  = (const float*)d_in[10];
  const float* Wf  = (const float*)d_in[11];
  const float* bf  = (const float*)d_in[12];

  char* ws = (char*)d_ws;
  const size_t MB = 1024u * 1024u;
  __bf16* xq  = (__bf16*)(ws + 0 * MB);    // [4096,1024] bf16   8 MB
  __bf16* xk  = (__bf16*)(ws + 8 * MB);
  __bf16* xv  = (__bf16*)(ws + 16 * MB);
  __bf16* wtq = (__bf16*)(ws + 24 * MB);   // [1024,1024] bf16   2 MB each
  __bf16* wtk = (__bf16*)(ws + 26 * MB);
  __bf16* wtv = (__bf16*)(ws + 28 * MB);
  __bf16* wtf = (__bf16*)(ws + 30 * MB);
  __bf16* Qh  = (__bf16*)(ws + 32 * MB);   // [B,H,S,DH] bf16    8 MB
  __bf16* Kh  = (__bf16*)(ws + 40 * MB);   // [B,H,S,DH]
  __bf16* Vt  = (__bf16*)(ws + 48 * MB);   // [B,H,DH,S]
  __bf16* ctx = (__bf16*)(ws + 56 * MB);   // [B,S,DIM] bf16     8 MB

  cast_in_kernel<<<(MROWS * DIM_) / 256, 256, 0, stream>>>(q, k, v, xq, xk, xv);
  cast_w_kernel<<<(4 * DIM_ * DIM_) / 256, 256, 0, stream>>>(Wq, Wk, Wv, Wf,
                                                             wtq, wtk, wtv, wtf);

  gemm_wmma_kernel<0><<<128, 256, 0, stream>>>(xq, wtq, bq, (void*)Qh);
  gemm_wmma_kernel<0><<<128, 256, 0, stream>>>(xk, wtk, bk, (void*)Kh);
  gemm_wmma_kernel<1><<<128, 256, 0, stream>>>(xv, wtv, bv, (void*)Vt);

  attn_kernel<<<B_ * H_ * (S_ / 16), 32, 0, stream>>>(Qh, Kh, Vt, pad, train, ctx);

  gemm_wmma_kernel<2><<<128, 256, 0, stream>>>(ctx, wtf, bf, d_out);
}
